// ImprovedGRU_42322607735489
// MI455X (gfx1250) — matrix operand order repaired and checked by reading, not compiled, not run on hardware
//
#include <hip/hip_runtime.h>
#include <hip/hip_bf16.h>

// ---------------------------------------------------------------------------
// ImprovedGRU for MI455X (gfx1250), wave32 + WMMA bf16 + TDM async tiles.
//   cvt_weights : W/U f32 -> bf16 workspace (L2-resident).
//   proj_kernel : Pz/Pa/Ph[t][b][h] = x @ W^T (bf16 WMMA, f32 acc), x tile in
//                 LDS shared across the three weight matrices.
//   scan_kernel : sequential recurrence; 16 WGs x 16 rows; h in LDS; U
//                 fragments streamed from L2 via global_load_b128; per-step
//                 projection tiles DMA'd into LDS one step ahead with
//                 TENSOR_LOAD_TO_LDS (double-buffered, s_wait_tensorcnt).
// Workspace: 3*T*B*H bf16 (192MB) + 6*H*H bf16 (~0.8MB)  ->  ~202 MB.
// ---------------------------------------------------------------------------

typedef __bf16 bf16;
typedef __attribute__((ext_vector_type(16))) __bf16 v16bf;
typedef __attribute__((ext_vector_type(8)))  __bf16 bf16x8;
typedef __attribute__((ext_vector_type(8)))  float    v8f;
typedef __attribute__((ext_vector_type(4)))  float    v4f;
typedef unsigned int v4u __attribute__((ext_vector_type(4)));
typedef int          v8i __attribute__((ext_vector_type(8)));
typedef int          v4i __attribute__((ext_vector_type(4)));

#define WMMA_BF16(A_, B_, C_) \
  __builtin_amdgcn_wmma_f32_16x16x32_bf16(false, (A_), false, (B_), (short)0, (C_), false, false)

constexpr int BB = 256;   // batch
constexpr int TT = 512;   // time
constexpr int II = 256;   // input dim
constexpr int HH = 256;   // hidden dim

constexpr size_t PROJ   = (size_t)TT * BB * HH;
constexpr size_t OFF_PZ = 0;
constexpr size_t OFF_PA = PROJ;
constexpr size_t OFF_PH = 2 * PROJ;
constexpr size_t OFF_UZ = 3 * PROJ;
constexpr size_t OFF_UA = OFF_UZ + (size_t)HH * HH;
constexpr size_t OFF_UH = OFF_UA + (size_t)HH * HH;
constexpr size_t OFF_WZ = OFF_UH + (size_t)HH * HH;
constexpr size_t OFF_WA = OFF_WZ + (size_t)HH * II;
constexpr size_t OFF_WH = OFF_WA + (size_t)HH * II;

// --- explicit global-address-space access: emit global_load/store_* (not flat)
// NOTE: only use with scalar / ext_vector types (trivially loadable across AS).
template <typename T>
__device__ __forceinline__ T gload(const void* p) {
  return *(const __attribute__((address_space(1))) T*)(unsigned long long)p;
}
template <typename T>
__device__ __forceinline__ void gstore(void* p, T v) {
  *(__attribute__((address_space(1))) T*)(unsigned long long)p = v;
}

// --- fragment helpers (wave32; A/B use contiguous k = 16*(lane>>4) + s, not ISA table order) ----------------

// A 16x32 bf16 from row-major [16][256] LDS. lanes 0-15: K=kb+{0..15};
// lanes 16-31: K=kb+{16..31}.
__device__ __forceinline__ v16bf load_a_frag(const bf16* s, int kb) {
  const int lane = threadIdx.x & 31;
  const int row  = lane & 15;
  const int koff = (lane >> 4) * 16;
  bf16x8 x0 = *(const bf16x8*)(s + row * HH + kb + koff);
  bf16x8 x1 = *(const bf16x8*)(s + row * HH + kb + koff + 8);
  return __builtin_shufflevector(x0, x1, 0,1,2,3,4,5,6,7,8,9,10,11,12,13,14,15);
}

// B 32x16 bf16 from row-major U[N][K] global: lane n = nb+(lane&15),
// K = kb + 16*(lane>>4) + 0..15 contiguous (32B per lane).
__device__ __forceinline__ v16bf load_b_frag(const bf16* __restrict__ U, int nb, int kb) {
  const int lane = threadIdx.x & 31;
  const bf16* p = U + (size_t)(nb + (lane & 15)) * HH + kb + (lane >> 4) * 16;
  bf16x8 x0 = gload<bf16x8>(p);
  bf16x8 x1 = gload<bf16x8>(p + 8);
  return __builtin_shufflevector(x0, x1, 0,1,2,3,4,5,6,7,8,9,10,11,12,13,14,15);
}

// D/C 16x16 f32: n = nb+(lane&15), rows m = mb + 8*(lane>>4) + v.
__device__ __forceinline__ void store_d_f32(float* s, int mb, int nb, v8f d) {
  const int lane = threadIdx.x & 31;
  const int n = nb + (lane & 15);
  const int m = mb + (lane >> 4) * 8;
#pragma unroll
  for (int v = 0; v < 8; ++v) s[(m + v) * HH + n] = d[v];
}
__device__ __forceinline__ void store_d_bf16(bf16* s, int mb, int nb, v8f d) {
  const int lane = threadIdx.x & 31;
  const int n = nb + (lane & 15);
  const int m = mb + (lane >> 4) * 8;
#pragma unroll
  for (int v = 0; v < 8; ++v) s[(m + v) * HH + n] = (bf16)d[v];
}

__device__ __forceinline__ float fast_sigmoid(float x) {
  return 1.0f / (1.0f + __expf(-x));
}
__device__ __forceinline__ float fast_tanh(float x) {
  x = fminf(fmaxf(x, -15.0f), 15.0f);
  float e = __expf(-2.0f * x);
  return (1.0f - e) / (1.0f + e);
}

// --- TDM: 1-D contiguous 8KB tile (4096 x bf16) global -> LDS ---------------
// D# group0: count=1, lds_addr, 57b global addr, type=2.
// D# group1: data_size=2B, tensor_dim0=tile_dim0=4096, tensor_dim1=tile_dim1=1.
__device__ __forceinline__ void tdm_load_8k(unsigned lds_off, const void* gp) {
  unsigned long long ga = (unsigned long long)gp;
  v4u g0;
  g0[0] = 1u;                                            // count=1, user mode
  g0[1] = lds_off;                                       // lds_addr (bytes)
  g0[2] = (unsigned)(ga & 0xFFFFFFFFu);                  // global_addr[31:0]
  g0[3] = (unsigned)((ga >> 32) & 0x1FFFFFFu) | (2u << 30); // [56:32] | type=2
  v8i g1;
  g1[0] = 0x00010000;  // workgroup_mask=0, data_size=1 (2 bytes)
  g1[1] = 0x10000000;  // tensor_dim0[15:0]=4096 in [63:48]
  g1[2] = 0x00010000;  // tensor_dim0[31:16]=0, tensor_dim1[15:0]=1
  g1[3] = 0x10000000;  // tensor_dim1[31:16]=0, tile_dim0=4096
  g1[4] = 0x00000001;  // tile_dim1=1, tile_dim2=0
  g1[5] = 0x00001000;  // tensor_dim0_stride=4096
  g1[6] = 0;
  g1[7] = 0;
  v4i z4 = {0, 0, 0, 0};
#if __clang_major__ >= 23
  v8i z8 = {0, 0, 0, 0, 0, 0, 0, 0};
  __builtin_amdgcn_tensor_load_to_lds(g0, g1, z4, z4, z8, 0);
#else
  __builtin_amdgcn_tensor_load_to_lds(g0, g1, z4, z4, 0);
#endif
}

// --- Phase 1: weight conversion f32 -> bf16 ---------------------------------
__global__ __launch_bounds__(256) void cvt_weights(
    const float* __restrict__ Wz, const float* __restrict__ Uz,
    const float* __restrict__ Wa, const float* __restrict__ Ua,
    const float* __restrict__ Wh, const float* __restrict__ Uh,
    bf16* __restrict__ ws) {
  int i = blockIdx.x * 256 + threadIdx.x;
  if (i >= HH * II) return;
  gstore<bf16>(ws + OFF_WZ + i, (bf16)gload<float>(Wz + i));
  gstore<bf16>(ws + OFF_UZ + i, (bf16)gload<float>(Uz + i));
  gstore<bf16>(ws + OFF_WA + i, (bf16)gload<float>(Wa + i));
  gstore<bf16>(ws + OFF_UA + i, (bf16)gload<float>(Ua + i));
  gstore<bf16>(ws + OFF_WH + i, (bf16)gload<float>(Wh + i));
  gstore<bf16>(ws + OFF_UH + i, (bf16)gload<float>(Uh + i));
}

// --- Phase 2: input projections  P[t][b][n] = sum_k x[b][t][k] * W[n][k] ----
__global__ __launch_bounds__(256) void proj_kernel(
    const float* __restrict__ x, bf16* __restrict__ ws) {
  __shared__ bf16 xs[32 * HH];   // 16 KB x tile (bf16)
  __shared__ bf16 os[32 * HH];   // 16 KB output staging

  const int tid  = threadIdx.x;
  const int wave = tid >> 5;
  const int mrow0 = blockIdx.x * 32;           // M row base (single b: 32 | 512)
  const int b  = mrow0 >> 9;
  const int t0 = mrow0 & (TT - 1);

  {  // cooperative load+cvt of x block [32][256] f32 -> LDS bf16
    const int r = tid >> 3, c0 = (tid & 7) * 32;
    const float* gp = x + (size_t)(mrow0 + r) * II + c0;
#pragma unroll
    for (int j = 0; j < 8; ++j) {
      v4f v = gload<v4f>(gp + j * 4);
      bf16* d = &xs[r * HH + c0 + j * 4];
      d[0] = (bf16)v.x; d[1] = (bf16)v.y; d[2] = (bf16)v.z; d[3] = (bf16)v.w;
    }
  }
  __syncthreads();

  const int mb = (wave >> 2) * 16;
  const bf16* Wmat[3] = { ws + OFF_WZ, ws + OFF_WA, ws + OFF_WH };
  const size_t Poff[3] = { OFF_PZ, OFF_PA, OFF_PH };

  for (int j = 0; j < 3; ++j) {
    v8f acc[4] = {};
#pragma unroll
    for (int ks = 0; ks < 8; ++ks) {
      const int kb = ks * 32;
      v16bf a = load_a_frag(xs + mb * HH, kb);
#pragma unroll
      for (int q = 0; q < 4; ++q) {
        v16bf bv = load_b_frag(Wmat[j], ((wave & 3) * 4 + q) * 16, kb);
        acc[q] = WMMA_BF16(a, bv, acc[q]);
      }
    }
#pragma unroll
    for (int q = 0; q < 4; ++q)
      store_d_bf16(os, mb, ((wave & 3) * 4 + q) * 16, acc[q]);
    __syncthreads();
    {  // coalesced store to P[j] at [t][b][n]
      const int r = tid >> 3, c0 = (tid & 7) * 32;
      bf16* gp = ws + Poff[j] + ((size_t)(t0 + r) * BB + b) * HH + c0;
      const bf16* sp = &os[r * HH + c0];
#pragma unroll
      for (int u = 0; u < 4; ++u) gstore<bf16x8>(gp + u * 8, ((const bf16x8*)sp)[u]);
    }
    __syncthreads();
  }
}

// --- Phase 3: sequential scan ------------------------------------------------
// Dynamic-LDS layout (no static LDS -> offsets are descriptor-usable from 0):
constexpr unsigned SOFF_HF  = 0;                      // f32 [16][256]  16 KB
constexpr unsigned SOFF_GZ  = 16384;                  // f32 [16][256]  16 KB
constexpr unsigned SOFF_GA  = 32768;                  // f32 [16][256]  16 KB
constexpr unsigned SOFF_HB  = 49152;                  // bf16 [16][256]  8 KB
constexpr unsigned SOFF_HAB = 57344;                  // bf16 [16][256]  8 KB
constexpr unsigned SOFF_PT  = 65536;                  // bf16 [2][3][4096] 48 KB
constexpr unsigned SOFF_BZ  = SOFF_PT + 49152;        // f32 [256]
constexpr unsigned SOFF_VA  = SOFF_BZ + 1024;
constexpr unsigned SOFF_BH  = SOFF_VA + 1024;
constexpr unsigned SMEM_SCAN = SOFF_BH + 1024;        // 117760 B

__global__ __launch_bounds__(256) void scan_kernel(
    const bf16* __restrict__ ws,
    const float* __restrict__ bz, const float* __restrict__ va,
    const float* __restrict__ bh, float* __restrict__ out) {
  constexpr int R = 16;
  extern __shared__ char smem[];
  float* hf  = (float*)(smem + SOFF_HF);
  float* Gz  = (float*)(smem + SOFF_GZ);
  float* Ga  = (float*)(smem + SOFF_GA);
  bf16*  hb  = (bf16*)(smem + SOFF_HB);
  bf16*  hab = (bf16*)(smem + SOFF_HAB);
  bf16*  Pt  = (bf16*)(smem + SOFF_PT);
  float* bzs = (float*)(smem + SOFF_BZ);
  float* vas = (float*)(smem + SOFF_VA);
  float* bhs = (float*)(smem + SOFF_BH);

  const int tid  = threadIdx.x;
  const int wave = tid >> 5;
  const int b0   = blockIdx.x * R;
  const int r    = tid >> 4;                // elementwise row 0..15
  const int c0   = (tid & 15) * 16;         // 16 cols per thread
  const int nt0  = wave * 2;

  const bf16* Uz = ws + OFF_UZ;
  const bf16* Ua = ws + OFF_UA;
  const bf16* Uh = ws + OFF_UH;
  const bf16* Pg[3] = { ws + OFF_PZ, ws + OFF_PA, ws + OFF_PH };

  for (int i = tid; i < R * HH; i += 256) { hf[i] = 0.0f; hb[i] = (bf16)0.0f; }
  if (tid < HH) { bzs[tid] = bz[tid]; vas[tid] = va[tid]; bhs[tid] = bh[tid]; }
  if (wave == 0) {  // kick off tile 0 DMA before the loop
#pragma unroll
    for (int j = 0; j < 3; ++j)
      tdm_load_8k(SOFF_PT + j * 8192, Pg[j] + (size_t)b0 * HH);
  }
  __syncthreads();

  for (int t = 0; t < TT; ++t) {
    const unsigned cur = (unsigned)(t & 1) * 3 * 4096;

    if (wave == 0) {
      __builtin_amdgcn_s_wait_tensorcnt(0);       // tile t landed in LDS
      if (t + 1 < TT) {                           // DMA tile t+1 (other buffer)
        const unsigned nxt = (unsigned)((t + 1) & 1) * 24576;  // bytes
        const size_t goff = ((size_t)(t + 1) * BB + b0) * HH;
#pragma unroll
        for (int j = 0; j < 3; ++j)
          tdm_load_8k(SOFF_PT + nxt + j * 8192, Pg[j] + goff);
      }
    }

    // ---- phase A: Gz = h@Uz^T, Ga = h@Ua^T
    v8f az[2] = {}, aa[2] = {};
#pragma unroll
    for (int ks = 0; ks < 8; ++ks) {
      const int kb = ks * 32;
      v16bf a = load_a_frag(hb, kb);
#pragma unroll
      for (int q = 0; q < 2; ++q) {
        const int nb = (nt0 + q) * 16;
        v16bf bzf = load_b_frag(Uz, nb, kb);
        az[q] = WMMA_BF16(a, bzf, az[q]);
        v16bf baf = load_b_frag(Ua, nb, kb);
        aa[q] = WMMA_BF16(a, baf, aa[q]);
      }
    }
#pragma unroll
    for (int q = 0; q < 2; ++q) {
      store_d_f32(Gz, 0, (nt0 + q) * 16, az[q]);
      store_d_f32(Ga, 0, (nt0 + q) * 16, aa[q]);
    }
    __syncthreads();   // Gz/Ga ready AND tile t published to all waves

    // ---- elementwise 1: gate z, attention softmax, h_att
    const bf16* tz = Pt + cur + 0 * 4096 + r * HH + c0;
    const bf16* ta = Pt + cur + 1 * 4096 + r * HH + c0;
    float zv[16], av[16];
    float mx = -1e30f;
    {
      bf16x8 xz0 = *(const bf16x8*)tz,  xz1 = *(const bf16x8*)(tz + 8);
      bf16x8 xa0 = *(const bf16x8*)ta,  xa1 = *(const bf16x8*)(ta + 8);
#pragma unroll
      for (int i = 0; i < 16; ++i) {
        const int c = c0 + i;
        float xz = (float)(i < 8 ? xz0[i & 7] : xz1[i & 7]);
        float xa = (float)(i < 8 ? xa0[i & 7] : xa1[i & 7]);
        zv[i] = fast_sigmoid(xz + Gz[r * HH + c] + bzs[c]);
        float ai = fast_tanh(xa + Ga[r * HH + c]) * vas[c];
        av[i] = ai;
        mx = fmaxf(mx, ai);
      }
    }
#pragma unroll
    for (int msk = 1; msk < 16; msk <<= 1) mx = fmaxf(mx, __shfl_xor(mx, msk, 32));
    float sum = 0.0f;
#pragma unroll
    for (int i = 0; i < 16; ++i) { av[i] = __expf(av[i] - mx); sum += av[i]; }
#pragma unroll
    for (int msk = 1; msk < 16; msk <<= 1) sum += __shfl_xor(sum, msk, 32);
    const float inv = 1.0f / sum;
#pragma unroll
    for (int i = 0; i < 16; ++i) {
      const int c = c0 + i;
      hab[r * HH + c] = (bf16)(av[i] * inv * hf[r * HH + c]);
    }
    __syncthreads();

    // ---- phase B: Gh = h_att @ Uh^T  (reuse Gz buffer)
    v8f ah[2] = {};
#pragma unroll
    for (int ks = 0; ks < 8; ++ks) {
      const int kb = ks * 32;
      v16bf a = load_a_frag(hab, kb);
#pragma unroll
      for (int q = 0; q < 2; ++q) {
        v16bf bhf = load_b_frag(Uh, (nt0 + q) * 16, kb);
        ah[q] = WMMA_BF16(a, bhf, ah[q]);
      }
    }
#pragma unroll
    for (int q = 0; q < 2; ++q) store_d_f32(Gz, 0, (nt0 + q) * 16, ah[q]);
    __syncthreads();

    // ---- elementwise 2: candidate, blend, write h_new
    const bf16* th_ = Pt + cur + 2 * 4096 + r * HH + c0;
    bf16x8 xh0 = *(const bf16x8*)th_, xh1 = *(const bf16x8*)(th_ + 8);
    float hn[16];
#pragma unroll
    for (int i = 0; i < 16; ++i) {
      const int c = c0 + i;
      float xh = (float)(i < 8 ? xh0[i & 7] : xh1[i & 7]);
      float htil = fast_tanh(xh + Gz[r * HH + c] + bhs[c]);
      float hold = hf[r * HH + c];
      hn[i] = (1.0f - zv[i]) * hold + zv[i] * htil;
      hf[r * HH + c] = hn[i];            // exclusive (r,c) ownership
      hb[r * HH + c] = (bf16)hn[i];
    }
    {  // outputs[b][t][h]: 64B contiguous per thread -> global_store_b128 x4
      float* gp = out + ((size_t)(b0 + r) * TT + t) * HH + c0;
#pragma unroll
      for (int u = 0; u < 4; ++u) {
        v4f v = { hn[u*4+0], hn[u*4+1], hn[u*4+2], hn[u*4+3] };
        gstore<v4f>(gp + u * 4, v);
      }
    }
    __syncthreads();
  }

  // h_last -> out[B*T*H + b*H + c]
  {
    float* gp = out + (size_t)BB * TT * HH + (size_t)(b0 + r) * HH + c0;
    const float* sp = &hf[r * HH + c0];
#pragma unroll
    for (int u = 0; u < 4; ++u) {
      v4f v = { sp[u*4+0], sp[u*4+1], sp[u*4+2], sp[u*4+3] };
      gstore<v4f>(gp + u * 4, v);
    }
  }
}

// ---------------------------------------------------------------------------
extern "C" void kernel_launch(void* const* d_in, const int* in_sizes, int n_in,
                              void* d_out, int out_size, void* d_ws, size_t ws_size,
                              hipStream_t stream) {
  (void)in_sizes; (void)n_in; (void)out_size; (void)ws_size;
  const float* x  = (const float*)d_in[0];
  const float* Wz = (const float*)d_in[1];
  const float* Uz = (const float*)d_in[2];
  const float* bz = (const float*)d_in[3];
  const float* Wa = (const float*)d_in[4];
  const float* Ua = (const float*)d_in[5];
  const float* va = (const float*)d_in[6];
  const float* Wh = (const float*)d_in[7];
  const float* Uh = (const float*)d_in[8];
  const float* bh = (const float*)d_in[9];
  bf16*  ws  = (bf16*)d_ws;     // needs ~202 MB
  float* out = (float*)d_out;

  cvt_weights<<<dim3((HH * II + 255) / 256), dim3(256), 0, stream>>>(
      Wz, Uz, Wa, Ua, Wh, Uh, ws);
  proj_kernel<<<dim3((BB * TT) / 32), dim3(256), 0, stream>>>(x, ws);
  scan_kernel<<<dim3(BB / 16), dim3(256), SMEM_SCAN, stream>>>(ws, bz, va, bh, out);
}
